// Decoding_25769803776504
// MI455X (gfx1250) — matrix operand order repaired and checked by reading, not compiled, not run on hardware
//
#include <hip/hip_runtime.h>
#include <hip/hip_bf16.h>

typedef __attribute__((ext_vector_type(16))) _Float16 v16h;
typedef __attribute__((ext_vector_type(8)))  float    v8f;

#define NLAT  64
#define NCOMP 32
#define HALF_LOG_2PI 0.91893853320467274178f

// ---------------------------------------------------------------------------
// Pack latent (f32 [n_cells,64]) into WMMA A-matrix f16 layout.
// A tile (mt, kb): lane L holds 16 halfs:
//   m = mt*16 + (L&15)
//   k = kb*32 + ((L&16)?16:0) + j      j = 0..15
// Stored contiguously: out[((mt*2+kb)*32 + L)*16 + j]
// ---------------------------------------------------------------------------
__global__ void pack_latent_k(const float* __restrict__ latent,
                              _Float16* __restrict__ out, int total) {
    int o = blockIdx.x * blockDim.x + threadIdx.x;
    if (o >= total) return;
    int j    = o & 15;
    int lane = (o >> 4) & 31;
    int kb   = (o >> 9) & 1;
    int mt   = o >> 10;
    int m = mt * 16 + (lane & 15);
    int k = kb * 32 + ((lane & 16) ? 16 : 0) + j;
    out[o] = (_Float16)latent[m * NLAT + k];
}

// ---------------------------------------------------------------------------
// Pack gathered logit_weight[genes_oi[g]] (f32 [64,32]) into WMMA B layout.
// B tile (g, kb, ct): lane L holds 16 halfs:
//   k = kb*32 + ((L&16)?16:0) + j
//   c = ct*16 + (L&15)
// out[(((g*2+kb)*2+ct)*32 + L)*16 + j]
// ---------------------------------------------------------------------------
__global__ void pack_lw_k(const float* __restrict__ lw,
                          const int* __restrict__ genes_oi,
                          _Float16* __restrict__ out, int total) {
    int o = blockIdx.x * blockDim.x + threadIdx.x;
    if (o >= total) return;
    int j    = o & 15;
    int lane = (o >> 4) & 31;
    int ct   = (o >> 9) & 1;
    int kb   = (o >> 10) & 1;
    int g    = o >> 11;
    int k = kb * 32 + ((lane & 16) ? 16 : 0) + j;
    int c = ct * 16 + (lane & 15);
    long src = ((long)genes_oi[g] * NLAT + k) * NCOMP + c;
    out[o] = (_Float16)lw[src];
}

// ---------------------------------------------------------------------------
// Pack gathered rho_weight^T into WMMA B layout for the rho GEMM.
// B tile (nt, kb): lane L holds 16 halfs of W[k][n] = rho_weight[genes_oi[n],k]
//   n = nt*16 + (L&15),  k = kb*32 + ((L&16)?16:0) + j
// out[((nt*2+kb)*32 + L)*16 + j]
// ---------------------------------------------------------------------------
__global__ void pack_rw_k(const float* __restrict__ rw,
                          const int* __restrict__ genes_oi,
                          _Float16* __restrict__ out, int total) {
    int o = blockIdx.x * blockDim.x + threadIdx.x;
    if (o >= total) return;
    int j    = o & 15;
    int lane = (o >> 4) & 31;
    int kb   = (o >> 9) & 1;
    int nt   = o >> 10;
    int n = nt * 16 + (lane & 15);
    int k = kb * 32 + ((lane & 16) ? 16 : 0) + j;
    out[o] = (_Float16)rw[(long)genes_oi[n] * NLAT + k];
}

// ---------------------------------------------------------------------------
// mixture_delta[n,g,c] = sum_l latent[n,l] * logit_weight[genes_oi[g],l,c]
// One wave -> one 16x32 output tile (gene g, cell-tile mt): 2 n-tiles, K=64
// unrolled as 2 x v_wmma_f32_16x16x32_f16 per n-tile.
// ---------------------------------------------------------------------------
__global__ void wmma_delta_k(const _Float16* __restrict__ A,
                             const _Float16* __restrict__ B,
                             float* __restrict__ delta,
                             int n_genes, int nMT) {
    int wave = (int)((blockIdx.x * blockDim.x + threadIdx.x) >> 5);
    int lane = threadIdx.x & 31;
    int ntiles = n_genes * nMT;
    if (wave >= ntiles) return;          // wave-uniform branch: EXEC stays all-1s
    int g  = wave / nMT;
    int mt = wave - g * nMT;

    v8f acc0 = {}, acc1 = {};
#pragma unroll
    for (int kb = 0; kb < 2; ++kb) {
        v16h a  = *(const v16h*)(A + (((long)mt * 2 + kb) * 32 + lane) * 16);
        v16h b0 = *(const v16h*)(B + ((((long)g * 2 + kb) * 2 + 0) * 32 + lane) * 16);
        v16h b1 = *(const v16h*)(B + ((((long)g * 2 + kb) * 2 + 1) * 32 + lane) * 16);
        acc0 = __builtin_amdgcn_wmma_f32_16x16x32_f16(false, a, false, b0,
                                                      (short)0, acc0, false, false);
        acc1 = __builtin_amdgcn_wmma_f32_16x16x32_f16(false, a, false, b1,
                                                      (short)0, acc1, false, false);
    }
    // D layout: VGPR r -> M = r (+8 for lanes 16..31), N = lane&15
    int c0    = lane & 15;
    int mbase = mt * 16 + ((lane & 16) ? 8 : 0);
#pragma unroll
    for (int r = 0; r < 8; ++r) {
        long row = ((long)(mbase + r) * n_genes + g) * NCOMP;
        delta[row + c0]      = acc0[r];
        delta[row + 16 + c0] = acc1[r];
    }
}

// ---------------------------------------------------------------------------
// rho[n,g] = sum_l latent[n,l] * rho_weight[genes_oi[g],l]
// ---------------------------------------------------------------------------
__global__ void wmma_rho_k(const _Float16* __restrict__ A,
                           const _Float16* __restrict__ R,
                           float* __restrict__ rho,
                           int n_genes, int nMT, int nNT) {
    int wave = (int)((blockIdx.x * blockDim.x + threadIdx.x) >> 5);
    int lane = threadIdx.x & 31;
    int ntiles = nNT * nMT;
    if (wave >= ntiles) return;
    int nt = wave / nMT;
    int mt = wave - nt * nMT;

    v8f acc = {};
#pragma unroll
    for (int kb = 0; kb < 2; ++kb) {
        v16h a = *(const v16h*)(A + (((long)mt * 2 + kb) * 32 + lane) * 16);
        v16h b = *(const v16h*)(R + (((long)nt * 2 + kb) * 32 + lane) * 16);
        acc = __builtin_amdgcn_wmma_f32_16x16x32_f16(false, a, false, b,
                                                     (short)0, acc, false, false);
    }
    int ncol  = nt * 16 + (lane & 15);
    int mbase = mt * 16 + ((lane & 16) ? 8 : 0);
#pragma unroll
    for (int r = 0; r < 8; ++r)
        rho[(long)(mbase + r) * n_genes + ncol] = acc[r];
}

// ---------------------------------------------------------------------------
// Per-(local gene, comp) mixture parameters (hoisted out of the 1M-cut loop).
// ---------------------------------------------------------------------------
__global__ void gene_params_k(const float* __restrict__ loc_w,
                              const float* __restrict__ scale_w,
                              const float* __restrict__ logit_w,
                              const int* __restrict__ genes_oi,
                              float* __restrict__ loc, float* __restrict__ rscale,
                              float* __restrict__ lsc, float* __restrict__ lgw,
                              int total) {
    int idx = blockIdx.x * blockDim.x + threadIdx.x;
    if (idx >= total) return;
    int g = idx >> 5, c = idx & 31;
    long s = (long)genes_oi[g] * NCOMP + c;
    loc[idx]    = 1.0f / (1.0f + __expf(-loc_w[s]));
    float sc    = 1e-5f + __expf(scale_w[s]);
    rscale[idx] = 1.0f / sc;
    lsc[idx]    = __logf(sc);
    lgw[idx]    = logit_w[s];
}

// ---------------------------------------------------------------------------
// Zero counts + the two double accumulators (harness does not re-poison).
// ---------------------------------------------------------------------------
__global__ void zero_k(float* __restrict__ counts, double* __restrict__ acc, int n) {
    int i = blockIdx.x * blockDim.x + threadIdx.x;
    if (i < n) counts[i] = 0.0f;
    if (i < 2) acc[i] = 0.0;
}

__global__ void scatter_counts_k(const int* __restrict__ ix,
                                 float* __restrict__ counts, int n) {
    int i = blockIdx.x * blockDim.x + threadIdx.x;
    if (i < n) atomicAdd(&counts[ix[i]], 1.0f);
}

// ---------------------------------------------------------------------------
// Per-cut mixture log-likelihood:
//   LSE_c(logits+comp_lp) - LSE_c(logits), online (streaming) logsumexp.
// Block-reduce into a double accumulator.
// ---------------------------------------------------------------------------
__global__ void mixture_ll_k(const float* __restrict__ coords,
                             const int* __restrict__ cut_cxg,
                             const int* __restrict__ cut_g,
                             const float* __restrict__ delta,
                             const float* __restrict__ loc,
                             const float* __restrict__ rscale,
                             const float* __restrict__ lsc,
                             const float* __restrict__ lgw,
                             double* __restrict__ acc, int n_cuts) {
    int k = blockIdx.x * blockDim.x + threadIdx.x;
    float lm = 0.0f;
    if (k < n_cuts) {
        int   g = cut_g[k];
        float x = coords[k];
        long pbase = (long)g * NCOMP;
        long dbase = (long)cut_cxg[k] * NCOMP;
        float m1 = -__builtin_inff(), s1 = 0.0f;
        float m2 = -__builtin_inff(), s2 = 0.0f;
#pragma unroll 4
        for (int c = 0; c < NCOMP; ++c) {
            float logit = lgw[pbase + c] + delta[dbase + c];
            float z = (x - loc[pbase + c]) * rscale[pbase + c];
            float a = logit - 0.5f * z * z - lsc[pbase + c] - HALF_LOG_2PI;
            if (a > m1)     { s1 = s1 * __expf(m1 - a) + 1.0f; m1 = a; }
            else            { s1 += __expf(a - m1); }
            if (logit > m2) { s2 = s2 * __expf(m2 - logit) + 1.0f; m2 = logit; }
            else            { s2 += __expf(logit - m2); }
        }
        lm = (m1 + __logf(s1)) - (m2 + __logf(s2));
    }
    __shared__ float red[256];
    red[threadIdx.x] = lm;
    __syncthreads();
    for (int s = 128; s > 0; s >>= 1) {
        if (threadIdx.x < (unsigned)s) red[threadIdx.x] += red[threadIdx.x + s];
        __syncthreads();
    }
    if (threadIdx.x == 0) atomicAdd(&acc[0], (double)red[0]);
}

// ---------------------------------------------------------------------------
// Poisson fragment-count log-likelihood over n_cells x n_genes.
// ---------------------------------------------------------------------------
__global__ void frag_ll_k(const float* __restrict__ rho,
                          const float* __restrict__ counts,
                          const int* __restrict__ genes_oi,
                          const int* __restrict__ cells_oi,
                          const float* __restrict__ rho_bias,
                          const float* __restrict__ libsize,
                          double* __restrict__ acc, int n_cells, int n_genes) {
    int idx = blockIdx.x * blockDim.x + threadIdx.x;
    int total = n_cells * n_genes;
    float ll = 0.0f;
    if (idx < total) {
        int n = idx / n_genes;
        int g = idx - n * n_genes;
        float fe  = rho_bias[genes_oi[g]] * __expf(rho[idx]) * libsize[cells_oi[n]];
        float cnt = counts[idx];
        ll = cnt * __logf(fe) - fe - lgammaf(cnt + 1.0f);
    }
    __shared__ float red[256];
    red[threadIdx.x] = ll;
    __syncthreads();
    for (int s = 128; s > 0; s >>= 1) {
        if (threadIdx.x < (unsigned)s) red[threadIdx.x] += red[threadIdx.x + s];
        __syncthreads();
    }
    if (threadIdx.x == 0) atomicAdd(&acc[1], (double)red[0]);
}

__global__ void finalize_k(const double* __restrict__ acc, float* __restrict__ out) {
    if (blockIdx.x == 0 && threadIdx.x == 0)
        out[0] = (float)(-(acc[0] + acc[1]));
}

// ---------------------------------------------------------------------------
extern "C" void kernel_launch(void* const* d_in, const int* in_sizes, int n_in,
                              void* d_out, int out_size, void* d_ws, size_t ws_size,
                              hipStream_t stream) {
    const int*   local_cxg_ix = (const int*)  d_in[0];
    const float* coords       = (const float*)d_in[1];
    const float* latent       = (const float*)d_in[2];
    const int*   genes_oi     = (const int*)  d_in[3];
    const int*   cells_oi     = (const int*)  d_in[4];
    const int*   cut_cxg      = (const int*)  d_in[5];
    const int*   cut_g        = (const int*)  d_in[6];
    // d_in[7]=n_cells, d_in[8]=n_genes (device scalars) — derived from sizes instead:
    const float* loc_w        = (const float*)d_in[9];
    const float* scale_w      = (const float*)d_in[10];
    const float* logit_w      = (const float*)d_in[11];
    const float* logit_weight = (const float*)d_in[12];
    const float* rho_weight   = (const float*)d_in[13];
    const float* rho_bias     = (const float*)d_in[14];
    const float* libsize      = (const float*)d_in[15];

    const int n_frags = in_sizes[0];
    const int n_cuts  = in_sizes[1];
    const int n_cells = in_sizes[2] / NLAT;   // latent is [n_cells, 64]
    const int n_genes = in_sizes[3];          // genes_oi
    const int nMT = n_cells / 16;
    const int nNT = n_genes / 16;
    const long NG = (long)n_cells * n_genes;

    // ---- workspace layout (256B-aligned regions) ----
    char* base = (char*)d_ws;
    size_t off = 0;
    auto carve = [&](size_t bytes) { char* p = base + off;
                                     off += (bytes + 255) & ~(size_t)255; return p; };
    float*    delta  = (float*)   carve(NG * NCOMP * sizeof(float));
    float*    counts = (float*)   carve(NG * sizeof(float));
    float*    rho    = (float*)   carve(NG * sizeof(float));
    _Float16* packA  = (_Float16*)carve((size_t)nMT * 1024 * sizeof(_Float16));
    _Float16* packB  = (_Float16*)carve((size_t)n_genes * 2048 * sizeof(_Float16));
    _Float16* packR  = (_Float16*)carve((size_t)nNT * 1024 * sizeof(_Float16));
    float*    p_loc  = (float*)   carve((size_t)n_genes * NCOMP * sizeof(float));
    float*    p_rsc  = (float*)   carve((size_t)n_genes * NCOMP * sizeof(float));
    float*    p_lsc  = (float*)   carve((size_t)n_genes * NCOMP * sizeof(float));
    float*    p_lgw  = (float*)   carve((size_t)n_genes * NCOMP * sizeof(float));
    double*   acc    = (double*)  carve(2 * sizeof(double));

    const int T = 256;
    auto nb = [](long n, int t) { return (unsigned)((n + t - 1) / t); };

    // 1) zero counts + accumulators
    zero_k<<<nb(NG, T), T, 0, stream>>>(counts, acc, (int)NG);

    // 2) pack operands into WMMA layouts
    int totA = nMT * 1024;
    int totB = n_genes * 2048;
    int totR = nNT * 1024;
    pack_latent_k<<<nb(totA, T), T, 0, stream>>>(latent, packA, totA);
    pack_lw_k    <<<nb(totB, T), T, 0, stream>>>(logit_weight, genes_oi, packB, totB);
    pack_rw_k    <<<nb(totR, T), T, 0, stream>>>(rho_weight,   genes_oi, packR, totR);

    // 3) per-gene mixture params
    gene_params_k<<<nb((long)n_genes * NCOMP, T), T, 0, stream>>>(
        loc_w, scale_w, logit_w, genes_oi, p_loc, p_rsc, p_lsc, p_lgw,
        n_genes * NCOMP);

    // 4) WMMA GEMMs (8 waves / block, one 16x16(x2) tile per wave)
    long dThreads = (long)n_genes * nMT * 32;
    wmma_delta_k<<<nb(dThreads, T), T, 0, stream>>>(packA, packB, delta, n_genes, nMT);
    long rThreads = (long)nNT * nMT * 32;
    wmma_rho_k  <<<nb(rThreads, T), T, 0, stream>>>(packA, packR, rho, n_genes, nMT, nNT);

    // 5) fragment count scatter
    scatter_counts_k<<<nb(n_frags, T), T, 0, stream>>>(local_cxg_ix, counts, n_frags);

    // 6) per-cut mixture likelihood
    mixture_ll_k<<<nb(n_cuts, T), T, 0, stream>>>(
        coords, cut_cxg, cut_g, delta, p_loc, p_rsc, p_lsc, p_lgw, acc, n_cuts);

    // 7) Poisson fragment-count likelihood
    frag_ll_k<<<nb(NG, T), T, 0, stream>>>(
        rho, counts, genes_oi, cells_oi, rho_bias, libsize, acc, n_cells, n_genes);

    // 8) elbo = -(sum_mixture + sum_fragcounts)
    finalize_k<<<1, 1, 0, stream>>>(acc, (float*)d_out);
}
